// MultiHeadSelfAttention_21311627722867
// MI455X (gfx1250) — compile-verified, matrix operand order checked by reading, not run on hardware
//
#include <hip/hip_runtime.h>

// ---------------------------------------------------------------------------
// Multi-head self-attention for MI455X (gfx1250, wave32, WMMA bf16 path).
// B=4, S=2048, D=512, H=8, d_head=64.  Compute-bound -> everything through
// v_wmma_f32_16x16x32_bf16 with fp32 accumulation; flash-style online softmax.
// K/V tiles staged global->LDS with async-to-LDS (ASYNCcnt) when available.
// ---------------------------------------------------------------------------

constexpr int B_ = 4, S_ = 2048, D_ = 512, H_ = 8, DH_ = 64;
constexpr int M_ = B_ * S_;   // 8192 total rows

typedef __attribute__((ext_vector_type(16))) __bf16       bf16x16;
typedef __attribute__((ext_vector_type(8)))  float        f32x8;
typedef __attribute__((ext_vector_type(4)))  unsigned int u32x4;
typedef __attribute__((ext_vector_type(4)))  int          i32x4;

#if defined(__has_builtin)
#if __has_builtin(__builtin_amdgcn_global_load_async_to_lds_b128)
#define HAVE_ASYNC_LDS 1
#endif
#endif
#ifndef HAVE_ASYNC_LDS
#define HAVE_ASYNC_LDS 0
#endif

union Frag {
    bf16x16        v;
    u32x4          q[2];
    unsigned short s[16];
};

__device__ __forceinline__ unsigned short f2bf(float f) {
    union { float f; unsigned int u; } c; c.f = f;
    unsigned int u = c.u;
    u += 0x7FFFu + ((u >> 16) & 1u);   // round to nearest even
    return (unsigned short)(u >> 16);
}

// Load one 16-bit A/B WMMA fragment. `p` points at element [row_or_col][k0]
// of a K-contiguous row (16-byte aligned). Layout per CDNA5 ISA:
// elems 0..7 = K(8*half .. 8*half+7), elems 8..15 = K(16+8*half .. +7).
__device__ __forceinline__ bf16x16 load_frag(const unsigned short* p, int half) {
    Frag f;
    f.q[0] = *(const u32x4*)(p + 8 * half);
    f.q[1] = *(const u32x4*)(p + 16 + 8 * half);
    return f.v;
}

__device__ __forceinline__ f32x8 wmma_bf16(bf16x16 a, bf16x16 b, f32x8 c) {
    return __builtin_amdgcn_wmma_f32_16x16x32_bf16(
        /*neg_a=*/false, a, /*neg_b=*/false, b,
        /*c_mod=*/(short)0, c, /*reuse_a=*/false, /*reuse_b=*/false);
}

// 16-byte global -> LDS stage: async DMA path (ASYNCcnt) or register bounce.
// Builtin signature (from hipcc diagnostic): param0 = int4 addrspace(1)* (the
// global source, printed as 'int __device__ *'), 4 args total.
__device__ __forceinline__ void stage16(const unsigned short* src, unsigned short* dst) {
#if HAVE_ASYNC_LDS
    __builtin_amdgcn_global_load_async_to_lds_b128(
        (__attribute__((address_space(1))) i32x4*)src,
        (__attribute__((address_space(3))) i32x4*)dst,
        /*offset=*/0, /*cpol=*/0);
#else
    *(u32x4*)dst = *(const u32x4*)src;
#endif
}

__device__ __forceinline__ void wait_stage_and_sync() {
#if HAVE_ASYNC_LDS
    asm volatile("s_wait_asynccnt 0x0" ::: "memory");
#endif
    __syncthreads();
}

// ---------------------------------------------------------------------------
// Pre-pass: fp32 -> bf16 conversion (and weight transpose to [N][K]).
// ---------------------------------------------------------------------------
__global__ void cvt_bf16_kernel(const float* __restrict__ in,
                                unsigned short* __restrict__ out, int n) {
    int i = blockIdx.x * 256 + threadIdx.x;
    if (i < n) out[i] = f2bf(in[i]);
}

__global__ void cvt_bf16_T_kernel(const float* __restrict__ in,   // [rows][cols]
                                  unsigned short* __restrict__ out, // [cols][rows]
                                  int rows, int cols) {
    int i = blockIdx.x * 256 + threadIdx.x;
    if (i < rows * cols) {
        int r = i / cols, c = i % cols;
        out[c * rows + r] = f2bf(in[i]);
    }
}

// ---------------------------------------------------------------------------
// QKV projection: C[8192x512] = Xb @ W + bias.  WT is bf16 [N][K].
// mode 0: Q  -> scale 1/sqrt(64), store [b][h][s][d] bf16
// mode 1: K  -> store [b][h][s][d] bf16
// mode 2: V  -> store transposed [b][h][d][s] bf16 (packed b128 stores)
// Tile: workgroup = 256(M) x 64(N); wave = 32(M) x 64(N).
// 8 WMMA per K-step vs 12 b128 loads (B fragments reused across 2 A rows).
// ---------------------------------------------------------------------------
__global__ __launch_bounds__(256) void qkv_gemm_kernel(
    const unsigned short* __restrict__ X,   // [8192][512] bf16
    const unsigned short* __restrict__ WT,  // [512][512]  bf16 (N-major)
    const float* __restrict__ bias,         // [512]
    unsigned short* __restrict__ out, int mode) {
    const int lane = threadIdx.x & 31;
    const int wave = threadIdx.x >> 5;
    const int half = lane >> 4;
    const int l16  = lane & 15;
    const int m0 = blockIdx.x * 256 + wave * 32;
    const int n0 = blockIdx.y * 64;

    f32x8 acc[2][4] = {};
    const unsigned short* xrow0 = X + (size_t)(m0 + l16) * D_;
    const unsigned short* xrow1 = X + (size_t)(m0 + 16 + l16) * D_;

    for (int k0 = 0; k0 < D_; k0 += 32) {
        bf16x16 a0 = load_frag(xrow0 + k0, half);
        bf16x16 a1 = load_frag(xrow1 + k0, half);
#pragma unroll
        for (int t = 0; t < 4; t++) {
            const unsigned short* wrow = WT + (size_t)(n0 + t * 16 + l16) * D_ + k0;
            bf16x16 b = load_frag(wrow, half);
            acc[0][t] = wmma_bf16(a0, b, acc[0][t]);
            acc[1][t] = wmma_bf16(a1, b, acc[1][t]);
        }
    }

    const float scale = (mode == 0) ? 0.125f : 1.0f;  // 1/sqrt(64) folded into Q
#pragma unroll
    for (int mi = 0; mi < 2; mi++) {
        if (mode == 2) {
            // V^T layout [b][h][d][s]: lane's 8 rows (s) contiguous -> b128 store
#pragma unroll
            for (int t = 0; t < 4; t++) {
                int n = n0 + t * 16 + l16;            // column = h*64 + d
                int h = n >> 6, d = n & 63;
                float bv = bias[n];
                union { unsigned short s[8]; u32x4 v; } pk;
#pragma unroll
                for (int j = 0; j < 8; j++) pk.s[j] = f2bf(acc[mi][t][j] + bv);
                int m = m0 + mi * 16 + 8 * half;      // first row of the 8
                int b = m >> 11, s = m & 2047;
                size_t idx = (((size_t)b * H_ + h) * DH_ + d) * S_ + s;
                *(u32x4*)(out + idx) = pk.v;
            }
        } else {
#pragma unroll
            for (int t = 0; t < 4; t++) {
                int n = n0 + t * 16 + l16;
                int h = n >> 6, d = n & 63;
                float bv = bias[n];
#pragma unroll
                for (int j = 0; j < 8; j++) {
                    int m = m0 + mi * 16 + j + 8 * half;
                    int b = m >> 11, s = m & 2047;
                    size_t idx = (((size_t)b * H_ + h) * S_ + s) * DH_ + d;
                    out[idx] = f2bf((acc[mi][t][j] + bv) * scale);
                }
            }
        }
    }
}

// ---------------------------------------------------------------------------
// Flash attention. Grid: (S/128, B*H). 8 waves/WG, each wave owns 16 q-rows.
// K (32x64) and V^T (64x32) tiles staged global->LDS (double-buffered, one
// barrier per key tile); online softmax in fp32; P requantized to bf16
// through per-wave-private LDS (C-layout -> A-fragment layout).
// ---------------------------------------------------------------------------
__global__ __launch_bounds__(256) void flash_attn_kernel(
    const unsigned short* __restrict__ Q,   // [B][H][S][64] bf16, pre-scaled
    const unsigned short* __restrict__ K,   // [B][H][S][64] bf16
    const unsigned short* __restrict__ VT,  // [B][H][64][S] bf16
    unsigned short* __restrict__ Oc) {      // [B][S][512]   bf16 concat
    __shared__ unsigned short lk[2][32 * DH_];  // K tile  [key][d]   4KB x2
    __shared__ unsigned short lv[2][DH_ * 32];  // VT tile [d][key]   4KB x2
    __shared__ unsigned short lp[8 * 512];      // per-wave P staging 8KB

    const int tid  = threadIdx.x;
    const int lane = tid & 31;
    const int wave = tid >> 5;
    const int half = lane >> 4;
    const int l16  = lane & 15;
    const int bh   = blockIdx.y;            // b*8 + h
    const int q0   = blockIdx.x * 128 + wave * 16;

    const unsigned short* Qp = Q  + (size_t)bh * S_ * DH_;
    const unsigned short* Kp = K  + (size_t)bh * S_ * DH_;
    const unsigned short* Vp = VT + (size_t)bh * DH_ * S_;

    // Per-thread staging slots: K tile = 32 keys x 8 chunks of 16B,
    //                           V tile = 64 rows x 4 chunks of 16B.
    const int kkey = tid >> 3, kchk = tid & 7;
    const int vrow = tid >> 2, vchk = tid & 3;

    // Stage tile 0 into buffer 0.
    stage16(Kp + (size_t)kkey * DH_ + kchk * 8, &lk[0][kkey * DH_ + kchk * 8]);
    stage16(Vp + (size_t)vrow * S_ + vchk * 8,  &lv[0][vrow * 32 + vchk * 8]);

    const unsigned short* qrow = Qp + (size_t)(q0 + l16) * DH_;
    bf16x16 qf0 = load_frag(qrow + 0,  half);   // d = 0..31
    bf16x16 qf1 = load_frag(qrow + 32, half);   // d = 32..63

    f32x8 o[4] = {};
    float m_i[8], l_i[8];
#pragma unroll
    for (int j = 0; j < 8; j++) { m_i[j] = -1e30f; l_i[j] = 0.0f; }

    unsigned short* pbuf = lp + wave * 512;

    for (int kt = 0; kt < S_; kt += 32) {
        const int cur = (kt >> 5) & 1;

        wait_stage_and_sync();               // staged tile `cur` is ready

        if (kt + 32 < S_) {                  // stage next tile into other buffer
            const int nxt = cur ^ 1;
            stage16(Kp + (size_t)(kt + 32 + kkey) * DH_ + kchk * 8,
                    &lk[nxt][kkey * DH_ + kchk * 8]);
            stage16(Vp + (size_t)vrow * S_ + (kt + 32) + vchk * 8,
                    &lv[nxt][vrow * 32 + vchk * 8]);
        }

        // ---- scores: S = Q @ K^T (K tile rows are d-contiguous B frags) ----
        f32x8 s0 = {}, s1 = {};
        {
            const unsigned short* kr0 = &lk[cur][(l16)      * DH_];
            const unsigned short* kr1 = &lk[cur][(l16 + 16) * DH_];
            bf16x16 kb;
            kb = load_frag(kr0 + 0,  half); s0 = wmma_bf16(qf0, kb, s0);
            kb = load_frag(kr0 + 32, half); s0 = wmma_bf16(qf1, kb, s0);
            kb = load_frag(kr1 + 0,  half); s1 = wmma_bf16(qf0, kb, s1);
            kb = load_frag(kr1 + 32, half); s1 = wmma_bf16(qf1, kb, s1);
        }

        // ---- online softmax update (fp32) ----
        float p0[8], p1[8];
#pragma unroll
        for (int j = 0; j < 8; j++) {
            float tm = fmaxf(s0[j], s1[j]);
            tm = fmaxf(tm, __shfl_xor(tm, 1, 32));
            tm = fmaxf(tm, __shfl_xor(tm, 2, 32));
            tm = fmaxf(tm, __shfl_xor(tm, 4, 32));
            tm = fmaxf(tm, __shfl_xor(tm, 8, 32));
            float mn   = fmaxf(m_i[j], tm);
            float corr = __expf(m_i[j] - mn);
            m_i[j] = mn;
            p0[j] = __expf(s0[j] - mn);
            p1[j] = __expf(s1[j] - mn);
            float rs = p0[j] + p1[j];
            rs += __shfl_xor(rs, 1, 32);
            rs += __shfl_xor(rs, 2, 32);
            rs += __shfl_xor(rs, 4, 32);
            rs += __shfl_xor(rs, 8, 32);
            l_i[j] = l_i[j] * corr + rs;
            o[0][j] *= corr; o[1][j] *= corr; o[2][j] *= corr; o[3][j] *= corr;
        }

        // ---- P (C-layout fp32) -> LDS column-major [key][row], packed b128 ----
        union { unsigned short s[8]; u32x4 v; } t0, t1;
#pragma unroll
        for (int j = 0; j < 8; j++) { t0.s[j] = f2bf(p0[j]); t1.s[j] = f2bf(p1[j]); }
        *(u32x4*)(pbuf + (size_t)l16 * 16 + 8 * half)        = t0.v;
        *(u32x4*)(pbuf + (size_t)(l16 + 16) * 16 + 8 * half) = t1.v;

        // Same-wave LDS ops are in-order (DScnt); fence the compiler only.
        asm volatile("" ::: "memory");
        __builtin_amdgcn_wave_barrier();

        // ---- read back as bf16 A-fragment (row = l16, K = key) ----
        Frag pf;
#pragma unroll
        for (int e = 0; e < 8; e++) {
            pf.s[e]     = pbuf[(8 * half + e) * 16 + l16];
            pf.s[e + 8] = pbuf[(16 + 8 * half + e) * 16 + l16];
        }
        asm volatile("" ::: "memory");
        __builtin_amdgcn_wave_barrier();

        // ---- O += P @ V  (V^T tile rows are key-contiguous B fragments) ----
#pragma unroll
        for (int t = 0; t < 4; t++) {
            bf16x16 vb = load_frag(&lv[cur][(t * 16 + l16) * 32], half);
            o[t] = wmma_bf16(pf.v, vb, o[t]);
        }
    }

    // ---- epilogue: normalize and write concat [b][s][h*64+d] as bf16 ----
    const int b = bh >> 3, h = bh & 7;
#pragma unroll
    for (int t = 0; t < 4; t++) {
#pragma unroll
        for (int j = 0; j < 8; j++) {
            int   srow = q0 + j + 8 * half;
            int   dcol = h * 64 + t * 16 + l16;
            float val  = o[t][j] / l_i[j];
            Oc[((size_t)b * S_ + srow) * D_ + dcol] = f2bf(val);
        }
    }
}

// ---------------------------------------------------------------------------
// Output projection: out[8192x512] fp32 = Concat_bf16 @ Wh + bh.
// Same 256x64 workgroup tiling as the QKV GEMM.
// ---------------------------------------------------------------------------
__global__ __launch_bounds__(256) void out_gemm_kernel(
    const unsigned short* __restrict__ A,   // [8192][512] bf16
    const unsigned short* __restrict__ WT,  // [512][512]  bf16 (N-major)
    const float* __restrict__ bias,         // [512]
    float* __restrict__ out) {              // [8192][512] fp32
    const int lane = threadIdx.x & 31;
    const int wave = threadIdx.x >> 5;
    const int half = lane >> 4;
    const int l16  = lane & 15;
    const int m0 = blockIdx.x * 256 + wave * 32;
    const int n0 = blockIdx.y * 64;

    f32x8 acc[2][4] = {};
    const unsigned short* arow0 = A + (size_t)(m0 + l16) * D_;
    const unsigned short* arow1 = A + (size_t)(m0 + 16 + l16) * D_;

    for (int k0 = 0; k0 < D_; k0 += 32) {
        bf16x16 a0 = load_frag(arow0 + k0, half);
        bf16x16 a1 = load_frag(arow1 + k0, half);
#pragma unroll
        for (int t = 0; t < 4; t++) {
            const unsigned short* wrow = WT + (size_t)(n0 + t * 16 + l16) * D_ + k0;
            bf16x16 b = load_frag(wrow, half);
            acc[0][t] = wmma_bf16(a0, b, acc[0][t]);
            acc[1][t] = wmma_bf16(a1, b, acc[1][t]);
        }
    }

#pragma unroll
    for (int mi = 0; mi < 2; mi++) {
#pragma unroll
        for (int t = 0; t < 4; t++) {
            int   n  = n0 + t * 16 + l16;
            float bv = bias[n];
#pragma unroll
            for (int j = 0; j < 8; j++) {
                int m = m0 + mi * 16 + j + 8 * half;
                out[(size_t)m * D_ + n] = acc[mi][t][j] + bv;
            }
        }
    }
}

// ---------------------------------------------------------------------------
// Host-side launcher.
// Inputs: x, Wq, bq, Wk, bk, Wv, bv, Wh, bh  (all fp32)
// ---------------------------------------------------------------------------
extern "C" void kernel_launch(void* const* d_in, const int* in_sizes, int n_in,
                              void* d_out, int out_size, void* d_ws, size_t ws_size,
                              hipStream_t stream) {
    const float* x  = (const float*)d_in[0];
    const float* Wq = (const float*)d_in[1];
    const float* bq = (const float*)d_in[2];
    const float* Wk = (const float*)d_in[3];
    const float* bk = (const float*)d_in[4];
    const float* Wv = (const float*)d_in[5];
    const float* bv = (const float*)d_in[6];
    const float* Wh = (const float*)d_in[7];
    const float* bh = (const float*)d_in[8];
    float* out = (float*)d_out;

    // Workspace carve-up (bf16 = unsigned short), total ~42 MiB.
    unsigned short* xb  = (unsigned short*)d_ws;            // [8192][512]
    unsigned short* WqT = xb  + (size_t)M_ * D_;            // [512][512]
    unsigned short* WkT = WqT + (size_t)D_ * D_;
    unsigned short* WvT = WkT + (size_t)D_ * D_;
    unsigned short* WhT = WvT + (size_t)D_ * D_;
    unsigned short* Qb  = WhT + (size_t)D_ * D_;            // [B][H][S][64]
    unsigned short* Kb  = Qb  + (size_t)M_ * D_;            // [B][H][S][64]
    unsigned short* VTb = Kb  + (size_t)M_ * D_;            // [B][H][64][S]
    unsigned short* Ocb = VTb + (size_t)M_ * D_;            // [B][S][512]

    // 1) fp32 -> bf16 conversions (x direct, weights transposed to [N][K]).
    {
        int n = M_ * D_;
        cvt_bf16_kernel<<<(n + 255) / 256, 256, 0, stream>>>(x, xb, n);
        int nw = D_ * D_;
        dim3 g((nw + 255) / 256);
        cvt_bf16_T_kernel<<<g, 256, 0, stream>>>(Wq, WqT, D_, D_);
        cvt_bf16_T_kernel<<<g, 256, 0, stream>>>(Wk, WkT, D_, D_);
        cvt_bf16_T_kernel<<<g, 256, 0, stream>>>(Wv, WvT, D_, D_);
        cvt_bf16_T_kernel<<<g, 256, 0, stream>>>(Wh, WhT, D_, D_);
    }

    // 2) QKV projections (WMMA). Q pre-scaled by 1/sqrt(d_head); V transposed.
    {
        dim3 grid(M_ / 256, D_ / 64);
        qkv_gemm_kernel<<<grid, 256, 0, stream>>>(xb, WqT, bq, Qb, 0);
        qkv_gemm_kernel<<<grid, 256, 0, stream>>>(xb, WkT, bk, Kb, 1);
        qkv_gemm_kernel<<<grid, 256, 0, stream>>>(xb, WvT, bv, VTb, 2);
    }

    // 3) Flash attention (WMMA + async LDS staging + online softmax).
    {
        dim3 grid(S_ / 128, B_ * H_);
        flash_attn_kernel<<<grid, 256, 0, stream>>>(Qb, Kb, VTb, Ocb);
    }

    // 4) Output projection (WMMA, fp32 result).
    {
        dim3 grid(M_ / 256, D_ / 64);
        out_gemm_kernel<<<grid, 256, 0, stream>>>(Ocb, WhT, bh, out);
    }
}
